// EncoderLayer_11132555231236
// MI455X (gfx1250) — matrix operand order repaired and checked by reading, not compiled, not run on hardware
//
#include <hip/hip_runtime.h>
#include <hip/hip_bf16.h>

typedef __attribute__((ext_vector_type(16))) _Float16 v16h;
typedef __attribute__((ext_vector_type(8)))  float    v8f;

#define E_    1024
#define FF_   4096
#define HD_   64
#define H_    16
#define L_    2048   // CN*CL per batch
#define ROWS_ 4096   // B * L_

// ---------------- WMMA helpers (wave32, 16x16x32 f16 -> f32) ----------------

__device__ __forceinline__ v8f wmma16(v16h a, v16h b, v8f c) {
  return __builtin_amdgcn_wmma_f32_16x16x32_f16(false, a, false, b, (short)0, c, false, false);
}

// A fragment: 16x32 (MxK), source row-major, row stride ld (halves).
// Lane m = lane&15; lanes 0-15: K chunks [0..7],[16..23]; lanes 16-31: [8..15],[24..31].
__device__ __forceinline__ v16h load_a_frag(const _Float16* __restrict__ src, int ld) {
  int lane = threadIdx.x & 31;
  int m = lane & 15;
  int k0 = (lane < 16) ? 0 : 8;
  v16h a;
#pragma unroll
  for (int j = 0; j < 8; ++j) a[j]     = src[m * ld + k0 + j];
#pragma unroll
  for (int j = 0; j < 8; ++j) a[8 + j] = src[m * ld + k0 + 16 + j];
  return a;
}

// B fragment: 32x16 (KxN) given TRANSPOSED source Bt[n][k], row stride ld.
// Lane n = lane&15; lanes 0-15: K chunks [0..7],[16..23]; lanes 16-31: [8..15],[24..31].
__device__ __forceinline__ v16h load_bt_frag(const _Float16* __restrict__ srcT, int ld) {
  int lane = threadIdx.x & 31;
  int n = lane & 15;
  int k0 = (lane < 16) ? 0 : 8;
  v16h b;
#pragma unroll
  for (int j = 0; j < 8; ++j) b[j]     = srcT[n * ld + k0 + j];
#pragma unroll
  for (int j = 0; j < 8; ++j) b[8 + j] = srcT[n * ld + k0 + 16 + j];
  return b;
}

// ---------------- K0: weight transpose + f16 conversion ----------------

__global__ void k_prep(const float* __restrict__ Wo, const float* __restrict__ W1,
                       const float* __restrict__ W2,
                       _Float16* __restrict__ WoT, _Float16* __restrict__ W1T,
                       _Float16* __restrict__ W2T) {
  size_t gid = (size_t)blockIdx.x * 256 + threadIdx.x;
  if (gid < (size_t)E_ * E_) {            // Wo[k][n] -> WoT[n][k]
    int k = (int)(gid >> 10), n = (int)(gid & 1023);
    WoT[(size_t)n * E_ + k] = (_Float16)Wo[gid];
  }
  if (gid < (size_t)E_ * FF_) {
    { int e = (int)(gid >> 12), f = (int)(gid & 4095);   // W1[e][f] -> W1T[f][e]
      W1T[(size_t)f * E_ + e] = (_Float16)W1[gid]; }
    { int f = (int)(gid >> 10), e = (int)(gid & 1023);   // W2[f][e] -> W2T[e][f]
      W2T[(size_t)e * FF_ + f] = (_Float16)W2[gid]; }
  }
}

// ---------------- K1: per-head Q/K/V projections ----------------
// grid.x = s(2) * n(2) * h(16) * ltile(128) = 8192, 32 threads
// Q,K stored row-major [l][hd]; V stored TRANSPOSED per head [hd][l].

__global__ void k_proj(const float* __restrict__ value,
                       const float* __restrict__ Wq, const float* __restrict__ Wk,
                       const float* __restrict__ Wv,
                       _Float16* __restrict__ Qb, _Float16* __restrict__ Kb,
                       _Float16* __restrict__ Vt) {
  int t  = blockIdx.x;
  int lt = t & 127, h = (t >> 7) & 15, n = (t >> 11) & 1, s = t >> 12;
  int lane = threadIdx.x;

  __shared__ _Float16 Xs[16 * 64];
  __shared__ _Float16 WT[3][64 * 64];   // transposed: WT[w][ncol*64 + k]

  for (int i = lane; i < 64 * 64; i += 32) {
    int k = i >> 6, nc = i & 63;
    WT[0][nc * 64 + k] = (_Float16)Wq[i];
    WT[1][nc * 64 + k] = (_Float16)Wk[i];
    WT[2][nc * 64 + k] = (_Float16)Wv[i];
  }
  for (int i = lane; i < 16 * 64; i += 32) {
    int m = i >> 6, hd = i & 63;
    int l = lt * 16 + m, cn = l >> 7, cl = l & 127;
    size_t src = ((((size_t)n * 16 + cn) * 2 + s) * 128 + cl) * E_ + h * 64 + hd;
    Xs[i] = (_Float16)value[src];
  }
  __syncthreads();

  size_t headbase = ((size_t)(s * 2 + n) * H_ + h) * (size_t)L_ * HD_;
  size_t rowbase  = headbase + (size_t)lt * 16 * HD_;
  int nc = lane & 15, moff = (lane < 16) ? 0 : 8;

  for (int w = 0; w < 3; ++w) {
#pragma unroll
    for (int nt = 0; nt < 4; ++nt) {
      v8f acc = {};
#pragma unroll
      for (int ks = 0; ks < 2; ++ks) {
        v16h a = load_a_frag(Xs + ks * 32, 64);
        v16h b = load_bt_frag(&WT[w][nt * 16 * 64 + ks * 32], 64);
        acc = wmma16(a, b, acc);
      }
      if (w == 0) {
#pragma unroll
        for (int r = 0; r < 8; ++r)
          Qb[rowbase + (size_t)(r + moff) * HD_ + nt * 16 + nc] = (_Float16)acc[r];
      } else if (w == 1) {
#pragma unroll
        for (int r = 0; r < 8; ++r)
          Kb[rowbase + (size_t)(r + moff) * HD_ + nt * 16 + nc] = (_Float16)acc[r];
      } else {
        // transposed: Vt[headbase + col*L + l] ; per lane the 8 r's are contiguous l's
        size_t vb = headbase + (size_t)(nt * 16 + nc) * L_ + (size_t)lt * 16 + moff;
#pragma unroll
        for (int r = 0; r < 8; ++r)
          Vt[vb + r] = (_Float16)acc[r];
      }
    }
  }
}

// ---------------- K2: flash attention ----------------
// grid.x = s(2)*n(2)*h(16)*qtile(128) = 8192, 32 threads

__global__ void k_attn(const _Float16* __restrict__ Qg, const _Float16* __restrict__ Kg,
                       const _Float16* __restrict__ Vtg, _Float16* __restrict__ Og) {
  int t  = blockIdx.x;
  int qt = t & 127, h = (t >> 7) & 15, n = (t >> 11) & 1, s = t >> 12;
  int lane = threadIdx.x;
  const float SCALE = 0.03125f;   // 1/sqrt(E) = 1/32

  size_t hb = ((size_t)(s * 2 + n) * H_ + h) * (size_t)L_ * HD_;
  const _Float16* Qh = Qg + hb;
  const _Float16* Kh = Kg + hb;
  const _Float16* Vh = Vtg + hb;   // [hd][l] transposed

  __shared__ _Float16 Ps[16 * 32];

  v16h aq0 = load_a_frag(Qh + (size_t)qt * 16 * HD_ + 0,  HD_);
  v16h aq1 = load_a_frag(Qh + (size_t)qt * 16 * HD_ + 32, HD_);

  float mrow[8], lrow[8];
  v8f o[4] = {};
#pragma unroll
  for (int r = 0; r < 8; ++r) { mrow[r] = -1e30f; lrow[r] = 0.f; }

  int nc = lane & 15, moff = (lane < 16) ? 0 : 8;

  for (int j = 0; j < L_ / 32; ++j) {
    int kb = j * 32;
    if (j + 1 < L_ / 32) {   // prefetch next K tile / V columns -> global_prefetch_b8
      __builtin_prefetch(Kh + (size_t)(kb + 32) * HD_, 0, 3);
      __builtin_prefetch(Vh + (size_t)(lane & 15) * L_ + kb + 32, 0, 3);
    }
    v8f s0 = {}, s1 = {};
    {
      v16h b0 = load_bt_frag(Kh + (size_t)kb * HD_ + 0,  HD_);
      v16h b1 = load_bt_frag(Kh + (size_t)kb * HD_ + 32, HD_);
      s0 = wmma16(aq0, b0, s0);
      s0 = wmma16(aq1, b1, s0);
    }
    {
      v16h b0 = load_bt_frag(Kh + (size_t)(kb + 16) * HD_ + 0,  HD_);
      v16h b1 = load_bt_frag(Kh + (size_t)(kb + 16) * HD_ + 32, HD_);
      s1 = wmma16(aq0, b0, s1);
      s1 = wmma16(aq1, b1, s1);
    }
    // online softmax over 32 columns (cols live across 16-lane half-wave)
#pragma unroll
    for (int r = 0; r < 8; ++r) {
      float a0 = s0[r] * SCALE, a1 = s1[r] * SCALE;
      float tmax = fmaxf(a0, a1);
#pragma unroll
      for (int msk = 8; msk >= 1; msk >>= 1)
        tmax = fmaxf(tmax, __shfl_xor(tmax, msk, 32));
      float nm   = fmaxf(mrow[r], tmax);
      float corr = __expf(mrow[r] - nm);
      float p0 = __expf(a0 - nm), p1 = __expf(a1 - nm);
      float rs = p0 + p1;
#pragma unroll
      for (int msk = 8; msk >= 1; msk >>= 1)
        rs += __shfl_xor(rs, msk, 32);
      lrow[r] = lrow[r] * corr + rs;
      mrow[r] = nm;
#pragma unroll
      for (int q4 = 0; q4 < 4; ++q4) o[q4][r] *= corr;
      int m = r + moff;
      Ps[m * 32 + nc]      = (_Float16)p0;
      Ps[m * 32 + 16 + nc] = (_Float16)p1;
    }
    __syncthreads();
    v16h ap = load_a_frag(Ps, 32);
    __syncthreads();
#pragma unroll
    for (int q4 = 0; q4 < 4; ++q4) {
      // B = V[kb..kb+31][q4*16..+16]; from transposed Vt rows are contiguous in k
      v16h bv = load_bt_frag(Vh + (size_t)(q4 * 16) * L_ + kb, L_);
      o[q4] = wmma16(ap, bv, o[q4]);
    }
  }

  size_t obase = (size_t)s * ROWS_ * E_ + ((size_t)n * L_ + (size_t)qt * 16) * E_ + h * 64;
#pragma unroll
  for (int r = 0; r < 8; ++r) {
    float inv = 1.f / lrow[r];
    int m = r + moff;
#pragma unroll
    for (int q4 = 0; q4 < 4; ++q4)
      Og[obase + (size_t)m * E_ + q4 * 16 + nc] = (_Float16)(o[q4][r] * inv);
  }
}

// ---------------- K3: (O0+O1) @ WoT + 2*bo + query -> Y (f32) ----------------
// grid (128, 16), 32 threads; 32-row x 64-col tile per wave

__global__ void k_wo(const _Float16* __restrict__ O, const _Float16* __restrict__ WoT,
                     const float* __restrict__ bo, const float* __restrict__ query,
                     float* __restrict__ Y) {
  int rt = blockIdx.x, ct = blockIdx.y, lane = threadIdx.x;
  v8f acc[2][4] = {};
  const _Float16* A0 = O + (size_t)rt * 32 * E_;
  const _Float16* A1 = A0 + (size_t)ROWS_ * E_;
  for (int ks = 0; ks < E_ / 32; ++ks) {
    v16h aL = load_a_frag(A0 + ks * 32, E_) + load_a_frag(A1 + ks * 32, E_);
    v16h aH = load_a_frag(A0 + 16 * E_ + ks * 32, E_) + load_a_frag(A1 + 16 * E_ + ks * 32, E_);
#pragma unroll
    for (int t4 = 0; t4 < 4; ++t4) {
      v16h b = load_bt_frag(WoT + (size_t)(ct * 64 + t4 * 16) * E_ + ks * 32, E_);
      acc[0][t4] = wmma16(aL, b, acc[0][t4]);
      acc[1][t4] = wmma16(aH, b, acc[1][t4]);
    }
  }
  int nc = lane & 15, moff = (lane < 16) ? 0 : 8;
#pragma unroll
  for (int mh = 0; mh < 2; ++mh)
#pragma unroll
    for (int t4 = 0; t4 < 4; ++t4)
#pragma unroll
      for (int r = 0; r < 8; ++r) {
        int col = ct * 64 + t4 * 16 + nc;
        size_t row = (size_t)rt * 32 + mh * 16 + r + moff;
        Y[row * E_ + col] = acc[mh][t4][r] + 2.f * bo[col] + query[row * E_ + col];
      }
}

// ---------------- K4/K7: row LayerNorm over E=1024 ----------------
// grid 4096, 256 threads

__global__ void k_ln(const float* __restrict__ Yin, const float* __restrict__ g,
                     const float* __restrict__ b, float* __restrict__ Xout,
                     _Float16* __restrict__ X16) {
  __shared__ float red[256];
  int row = blockIdx.x, tid = threadIdx.x;
  const float* y = Yin + (size_t)row * E_;
  float v0[4]; float s = 0.f, ss = 0.f;
#pragma unroll
  for (int i = 0; i < 4; ++i) { float t = y[tid + i * 256]; v0[i] = t; s += t; ss += t * t; }
  red[tid] = s; __syncthreads();
  for (int st = 128; st > 0; st >>= 1) { if (tid < st) red[tid] += red[tid + st]; __syncthreads(); }
  float mean = red[0] * (1.f / E_); __syncthreads();
  red[tid] = ss; __syncthreads();
  for (int st = 128; st > 0; st >>= 1) { if (tid < st) red[tid] += red[tid + st]; __syncthreads(); }
  float var  = red[0] * (1.f / E_) - mean * mean;
  float rinv = rsqrtf(var + 1e-5f);
#pragma unroll
  for (int i = 0; i < 4; ++i) {
    int c = tid + i * 256;
    float xv = (v0[i] - mean) * rinv * g[c] + b[c];
    Xout[(size_t)row * E_ + c] = xv;
    if (X16) X16[(size_t)row * E_ + c] = (_Float16)xv;
  }
}

// ---------------- K5: H = relu(X @ W1T + bf1) (f16) ----------------
// grid (128, 64), 32 threads; 32x64 tile per wave

__global__ void k_ffn1(const _Float16* __restrict__ X16, const _Float16* __restrict__ W1T,
                       const float* __restrict__ bf1, _Float16* __restrict__ Hh) {
  int rt = blockIdx.x, ct = blockIdx.y, lane = threadIdx.x;
  v8f acc[2][4] = {};
  const _Float16* A = X16 + (size_t)rt * 32 * E_;
  for (int ks = 0; ks < E_ / 32; ++ks) {
    v16h aL = load_a_frag(A + ks * 32, E_);
    v16h aH = load_a_frag(A + 16 * E_ + ks * 32, E_);
#pragma unroll
    for (int t4 = 0; t4 < 4; ++t4) {
      v16h b = load_bt_frag(W1T + (size_t)(ct * 64 + t4 * 16) * E_ + ks * 32, E_);
      acc[0][t4] = wmma16(aL, b, acc[0][t4]);
      acc[1][t4] = wmma16(aH, b, acc[1][t4]);
    }
  }
  int nc = lane & 15, moff = (lane < 16) ? 0 : 8;
#pragma unroll
  for (int mh = 0; mh < 2; ++mh)
#pragma unroll
    for (int t4 = 0; t4 < 4; ++t4)
#pragma unroll
      for (int r = 0; r < 8; ++r) {
        int col = ct * 64 + t4 * 16 + nc;
        size_t row = (size_t)rt * 32 + mh * 16 + r + moff;
        float hv = fmaxf(acc[mh][t4][r] + bf1[col], 0.f);
        Hh[row * FF_ + col] = (_Float16)hv;
      }
}

// ---------------- K6: Y2 = H @ W2T + bf2 + X (f32) ----------------
// grid (128, 16), 32 threads; 32x64 tile per wave

__global__ void k_ffn2(const _Float16* __restrict__ Hh, const _Float16* __restrict__ W2T,
                       const float* __restrict__ bf2, const float* __restrict__ Xf,
                       float* __restrict__ Y2) {
  int rt = blockIdx.x, ct = blockIdx.y, lane = threadIdx.x;
  v8f acc[2][4] = {};
  const _Float16* A = Hh + (size_t)rt * 32 * FF_;
  for (int ks = 0; ks < FF_ / 32; ++ks) {
    v16h aL = load_a_frag(A + ks * 32, FF_);
    v16h aH = load_a_frag(A + 16 * FF_ + ks * 32, FF_);
#pragma unroll
    for (int t4 = 0; t4 < 4; ++t4) {
      v16h b = load_bt_frag(W2T + (size_t)(ct * 64 + t4 * 16) * FF_ + ks * 32, FF_);
      acc[0][t4] = wmma16(aL, b, acc[0][t4]);
      acc[1][t4] = wmma16(aH, b, acc[1][t4]);
    }
  }
  int nc = lane & 15, moff = (lane < 16) ? 0 : 8;
#pragma unroll
  for (int mh = 0; mh < 2; ++mh)
#pragma unroll
    for (int t4 = 0; t4 < 4; ++t4)
#pragma unroll
      for (int r = 0; r < 8; ++r) {
        int col = ct * 64 + t4 * 16 + nc;
        size_t row = (size_t)rt * 32 + mh * 16 + r + moff;
        Y2[row * E_ + col] = acc[mh][t4][r] + bf2[col] + Xf[row * E_ + col];
      }
}

// ---------------- launch ----------------

extern "C" void kernel_launch(void* const* d_in, const int* in_sizes, int n_in,
                              void* d_out, int out_size, void* d_ws, size_t ws_size,
                              hipStream_t stream) {
  const float* value = (const float*)d_in[0];
  const float* query = (const float*)d_in[2];
  const float* Wv    = (const float*)d_in[5];
  const float* Wk    = (const float*)d_in[6];
  const float* Wq    = (const float*)d_in[7];
  const float* Wo    = (const float*)d_in[8];
  const float* bo    = (const float*)d_in[9];
  const float* g1    = (const float*)d_in[10];
  const float* b1    = (const float*)d_in[11];
  const float* W1    = (const float*)d_in[12];
  const float* bf1   = (const float*)d_in[13];
  const float* W2    = (const float*)d_in[14];
  const float* bf2   = (const float*)d_in[15];
  const float* g3    = (const float*)d_in[16];
  const float* b3    = (const float*)d_in[17];

  char* ws = (char*)d_ws;
  const size_t MB = 1u << 20;
  _Float16* Qb  = (_Float16*)(ws + 0 * MB);     // 16 MB
  _Float16* Kb  = (_Float16*)(ws + 16 * MB);    // 16 MB
  _Float16* Vt  = (_Float16*)(ws + 32 * MB);    // 16 MB (per-head transposed)
  _Float16* Ob  = (_Float16*)(ws + 48 * MB);    // 16 MB (2 streams x 4096 x 1024)
  _Float16* Hh  = (_Float16*)(ws + 0 * MB);     // 32 MB, aliases Q/K (dead after k_wo)
  _Float16* WoT = (_Float16*)(ws + 64 * MB);    // 2 MB
  _Float16* W1T = (_Float16*)(ws + 66 * MB);    // 8 MB
  _Float16* W2T = (_Float16*)(ws + 74 * MB);    // 8 MB
  float*    Y   = (float*)(ws + 82 * MB);       // 16 MB (also reused as Y2)
  float*    Xf  = (float*)(ws + 98 * MB);       // 16 MB
  _Float16* X16 = (_Float16*)(ws + 114 * MB);   // 8 MB
  float*    out = (float*)d_out;

  k_prep<<<16384, 256, 0, stream>>>(Wo, W1, W2, WoT, W1T, W2T);
  k_proj<<<8192, 32, 0, stream>>>(value, Wq, Wk, Wv, Qb, Kb, Vt);
  k_attn<<<8192, 32, 0, stream>>>(Qb, Kb, Vt, Ob);
  k_wo<<<dim3(128, 16), 32, 0, stream>>>(Ob, WoT, bo, query, Y);
  k_ln<<<4096, 256, 0, stream>>>(Y, g1, b1, Xf, X16);
  k_ffn1<<<dim3(128, 64), 32, 0, stream>>>(X16, W1T, bf1, Hh);
  k_ffn2<<<dim3(128, 16), 32, 0, stream>>>(Hh, W2T, bf2, Xf, Y);
  k_ln<<<4096, 256, 0, stream>>>(Y, g3, b3, out, (_Float16*)nullptr);
}
